// WeightsLib2DAlpha_44478681317670
// MI455X (gfx1250) — matrix operand order repaired and checked by reading, not compiled, not run on hardware
//
#include <hip/hip_runtime.h>
#include <hip/hip_bf16.h>
#include <math.h>

#define EPSV 1e-12f
#define Hd 256
#define Wd 256
#define Bn 32
#define FF 8192
#define CTXD 1024

typedef __attribute__((ext_vector_type(16))) __bf16 v16bf;
typedef __attribute__((ext_vector_type(16))) float  v16f;
typedef __attribute__((ext_vector_type(8)))  float  v8f;

// manual RNE f32->bf16 (for the small one-time activation conversions)
__device__ __forceinline__ unsigned short f2bf_1(float x) {
  unsigned u = __float_as_uint(x);
  u += 0x7fffu + ((u >> 16) & 1u);
  return (unsigned short)(u >> 16);
}

// vector fptrunc: let the backend pick packed v_cvt_pk_bf16_f32 if present
__device__ __forceinline__ v16bf cvt16(v8f lo, v8f hi) {
  v16f full = __builtin_shufflevector(lo, hi, 0, 1, 2, 3, 4, 5, 6, 7,
                                      8, 9, 10, 11, 12, 13, 14, 15);
  return __builtin_convertvector(full, v16bf);
}

// ---- fast tanh (gfx1250 hardware TANH trans op when available) ------------
#if __has_builtin(__builtin_amdgcn_tanhf)
__device__ __forceinline__ float fast_tanh(float x) { return __builtin_amdgcn_tanhf(x); }
#else
__device__ __forceinline__ float fast_tanh(float x) { return tanhf(x); }
#endif

// ---------------------------------------------------------------------------
// GEMM: out[batch, n] = sum_k act[batch, k] * Wm[n, k] + bias[n]
// Weights: fp32 stream, converted in-register (A, 16x32 bf16 fragment).
// Activations: pre-converted bf16, B fragment = 1 contiguous 32B vector load.
// One 16n x 16batch WMMA tile per wave; 8 waves -> 64 n x 32 batch per block.
// grid.x = N/64, grid.y selects (Wa,ba,outa) vs (Wb,bb,outb).
// ---------------------------------------------------------------------------
__global__ __launch_bounds__(256) void gemm_wmma_bf16(
    const float* __restrict__ Wa, const float* __restrict__ ba, float* __restrict__ outa,
    const float* __restrict__ Wb, const float* __restrict__ bb, float* __restrict__ outb,
    const unsigned short* __restrict__ actbf, int N, int K)
{
  const float* Wm   = (blockIdx.y == 0) ? Wa : Wb;
  const float* bias = (blockIdx.y == 0) ? ba : bb;
  float*       out  = (blockIdx.y == 0) ? outa : outb;

  const int wave  = threadIdx.x >> 5;
  const int lane  = threadIdx.x & 31;
  const int l15   = lane & 15;
  const int khalf = lane >> 4;              // 0: lanes 0-15, 1: lanes 16-31
  const int nsub  = wave >> 1;              // 4 n-subtiles per block
  const int btile = wave & 1;               // batch 0-15 vs 16-31
  const int nbase = blockIdx.x * 64 + nsub * 16;

  // A (16x32 bf16): lane<16 holds K 0..15; lane>=16 holds K 16..31
  const float* wrow = Wm + (size_t)(nbase + l15) * K + 16 * khalf;
  // B (32x16 bf16): lanes 0-15 hold K 0..15, lanes 16-31 hold K 16..31 (contig bf16)
  const unsigned short* brow = actbf + (size_t)(btile * 16 + l15) * K + 16 * khalf;

  v8f acc = {};

#pragma unroll 2
  for (int k0 = 0; k0 < K; k0 += 32) {
    __builtin_prefetch(wrow + k0 + 512, 0, 0);   // global_prefetch weight stream

    v8f alo = *(const v8f*)(wrow + k0);          // K koff+0..7
    v8f ahi = *(const v8f*)(wrow + k0 + 8);      // K koff+8..15
    v16bf A = cvt16(alo, ahi);
    v16bf B = *(const v16bf*)(brow + k0);        // 16 contiguous bf16

    acc = __builtin_amdgcn_wmma_f32_16x16x32_bf16(false, A, false, B,
                                                  (short)0, acc, false, false);
  }

  // D layout: VGPR v -> (M = v + 8*khalf, N = l15)
#pragma unroll
  for (int v = 0; v < 8; ++v) {
    int n = nbase + v + 8 * khalf;
    out[(size_t)(btile * 16 + l15) * N + n] = acc[v] + bias[n];
  }
}

// ------------------------- reduction helpers -------------------------------
__device__ __forceinline__ float block_sum(float v, float* smem) {
  for (int o = 16; o > 0; o >>= 1) v += __shfl_xor(v, o, 32);
  int w = threadIdx.x >> 5;
  if ((threadIdx.x & 31) == 0) smem[w] = v;
  __syncthreads();
  if (threadIdx.x == 0) {
    float s = smem[0];
    for (int i = 1; i < 8; ++i) s += smem[i];
    smem[0] = s;
  }
  __syncthreads();
  float r = smem[0];
  __syncthreads();
  return r;
}

__device__ __forceinline__ float block_max(float v, float* smem) {
  for (int o = 16; o > 0; o >>= 1) v = fmaxf(v, __shfl_xor(v, o, 32));
  int w = threadIdx.x >> 5;
  if ((threadIdx.x & 31) == 0) smem[w] = v;
  __syncthreads();
  if (threadIdx.x == 0) {
    float s = smem[0];
    for (int i = 1; i < 8; ++i) s = fmaxf(s, smem[i]);
    smem[0] = s;
  }
  __syncthreads();
  float r = smem[0];
  __syncthreads();
  return r;
}

// ---- one-time f32 -> bf16 conversion of the dense input x -----------------
__global__ __launch_bounds__(256) void f32_to_bf16(const float* __restrict__ in,
                                                   unsigned short* __restrict__ out,
                                                   int n)
{
  int i = blockIdx.x * 256 + threadIdx.x;
  if (i < n) out[i] = f2bf_1(in[i]);
}

// ---- norm over 2048-groups of xt/xg (ddof=1) + tanh gate -> bf16 ctx ------
__global__ __launch_bounds__(256) void norm_gate(const float* __restrict__ xt,
                                                 const float* __restrict__ xg,
                                                 unsigned short* __restrict__ ctxbf)
{
  __shared__ float sm[8];
  size_t base = (size_t)blockIdx.y * FF + (size_t)blockIdx.x * 2048;
  float vt[8], vg[8];
  float st = 0.f, sst = 0.f, sg = 0.f, ssg = 0.f;
#pragma unroll
  for (int i = 0; i < 8; ++i) {
    int idx = threadIdx.x + i * 256;
    vt[i] = xt[base + idx]; vg[i] = xg[base + idx];
    st += vt[i]; sst += vt[i] * vt[i];
    sg += vg[i]; ssg += vg[i] * vg[i];
  }
  st  = block_sum(st,  sm);
  sst = block_sum(sst, sm);
  sg  = block_sum(sg,  sm);
  ssg = block_sum(ssg, sm);
  const float n = 2048.0f;
  float mt = st / n, mg = sg / n;
  float rt = rsqrtf((sst - n * mt * mt) / (n - 1.0f) + EPSV);
  float rg = rsqrtf((ssg - n * mg * mg) / (n - 1.0f) + EPSV);
#pragma unroll
  for (int i = 0; i < 8; ++i) {
    int idx = threadIdx.x + i * 256;
    float c = ((vt[i] - mt) * rt) * fast_tanh((vg[i] - mg) * rg);
    ctxbf[base + idx] = f2bf_1(c);
  }
}

// ---- norm over 512-groups (ddof=1) -> modc --------------------------------
__global__ __launch_bounds__(256) void norm512(const float* __restrict__ in,
                                               float* __restrict__ outm)
{
  __shared__ float sm[8];
  size_t base = (size_t)blockIdx.y * FF + (size_t)blockIdx.x * 512;
  float v0 = in[base + threadIdx.x];
  float v1 = in[base + threadIdx.x + 256];
  float s  = block_sum(v0 + v1, sm);
  float ss = block_sum(v0 * v0 + v1 * v1, sm);
  const float n = 512.0f;
  float m = s / n;
  float r = rsqrtf((ss - n * m * m) / (n - 1.0f) + EPSV);
  outm[base + threadIdx.x]       = (v0 - m) * r;
  outm[base + threadIdx.x + 256] = (v1 - m) * r;
}

// ---- pass 1: fused tanh outer product, rank conv/sum, complex mix, maxes --
__global__ __launch_bounds__(256) void pixel_pass1(
    const float* __restrict__ modc, const float* __restrict__ conv_w,
    const float* __restrict__ conv_b, const float* __restrict__ wstat,
    const float* __restrict__ wmod, float* __restrict__ ms,
    float* __restrict__ proj, float* __restrict__ red)
{
  __shared__ float u[16];
  __shared__ float sm[8];
  int b = blockIdx.y, i = blockIdx.x, j = threadIdx.x;
  if (threadIdx.x < 16) {
    int k = threadIdx.x >> 2, r = threadIdx.x & 3;
    u[threadIdx.x] = modc[(size_t)b * FF + k * 2048 + r * 512 + i];
  }
  __syncthreads();
  // conv over rank then sum over out-channels collapses to per-rank weights
  float cw[4];
#pragma unroll
  for (int r = 0; r < 4; ++r)
    cw[r] = conv_w[r] + conv_w[4 + r] + conv_w[8 + r] + conv_w[12 + r];
  float cb = conv_b[0] + conv_b[1] + conv_b[2] + conv_b[3];

  float ch[4];
#pragma unroll
  for (int k = 0; k < 4; ++k) {
    float acc = 0.f;
#pragma unroll
    for (int r = 0; r < 4; ++r) {
      float vv = modc[(size_t)b * FF + k * 2048 + r * 512 + 256 + j];
      float t  = fast_tanh(u[k * 4 + r] * vv);
      acc += (k == 0) ? cw[r] * t : t;
    }
    ch[k] = acc;
  }
  ch[0] += cb;

  int pix = i * Wd + j;
  float m0 = wmod[0 * Hd * Wd + pix] * ch[0];
  float m1 = wmod[1 * Hd * Wd + pix] * ch[1];
  float m2 = wmod[2 * Hd * Wd + pix] * ch[2];
  float m3 = wmod[3 * Hd * Wd + pix] * ch[3];

  float A1r = wstat[(size_t)(Hd * Wd + pix) * 2 + 0];
  float A1i = wstat[(size_t)(Hd * Wd + pix) * 2 + 1];
  float msr = A1r * m0 - A1i * m1;
  float msi = A1r * m1 - A1i * m0;

  size_t gp = (size_t)b * Hd * Wd + pix;
  ms[2 * gp]       = msr; ms[2 * gp + 1]   = msi;
  proj[2 * gp]     = m2;  proj[2 * gp + 1] = m3;

  float a   = sqrtf(msr * msr + msi * msi + EPSV);
  float ap  = sqrtf(m2 * m2 + m3 * m3 + EPSV);
  float am  = block_max(a,  sm);
  float apm = block_max(ap, sm);
  if (threadIdx.x == 0) {
    atomicMax((int*)&red[b],      __float_as_int(am));   // positive floats
    atomicMax((int*)&red[Bn + b], __float_as_int(apm));
  }
}

// ---- pass 2: per-batch mean of post-polar magnitude -----------------------
__global__ __launch_bounds__(256) void pixel_pass2(const float* __restrict__ ms,
                                                   float* __restrict__ red)
{
  __shared__ float sm[8];
  int b = blockIdx.y;
  size_t gp = (size_t)b * Hd * Wd + blockIdx.x * Wd + threadIdx.x;
  float msr = ms[2 * gp], msi = ms[2 * gp + 1];
  float a  = sqrtf(msr * msr + msi * msi + EPSV);
  float h  = a / (red[b] + EPSV);
  float s0 = h * msr / (a + EPSV);
  float s1 = h * msi / (a + EPSV);
  float mp = sqrtf(s0 * s0 + s1 * s1 + EPSV);
  float tot = block_sum(mp, sm);
  if (threadIdx.x == 0) atomicAdd(&red[2 * Bn + b], tot);
}

// ---- pass 3: final weights ------------------------------------------------
__global__ __launch_bounds__(256) void pixel_pass3(
    const float* __restrict__ ms, const float* __restrict__ proj,
    const float* __restrict__ wstat, const float* __restrict__ red,
    float* __restrict__ out)
{
  int b = blockIdx.y, i = blockIdx.x, j = threadIdx.x;
  int pix = i * Wd + j;
  size_t gp = (size_t)b * Hd * Wd + pix;

  float msr = ms[2 * gp], msi = ms[2 * gp + 1];
  float a  = sqrtf(msr * msr + msi * msi + EPSV);
  float h  = a / (red[b] + EPSV);
  float s0 = h * msr / (a + EPSV);
  float s1 = h * msi / (a + EPSV);
  float magb = sqrtf(red[2 * Bn + b] / (float)(Hd * Wd) + EPSV);
  s0 /= magb; s1 /= magb;
  float w0 = wstat[(size_t)pix * 2]     + s0;
  float w1 = wstat[(size_t)pix * 2 + 1] + s1;

  float p0 = proj[2 * gp], p1 = proj[2 * gp + 1];
  float apx = sqrtf(p0 * p0 + p1 * p1 + EPSV);
  float hp  = apx / (red[Bn + b] + EPSV);
  float q0  = hp * p0 / (apx + EPSV);
  float q1  = hp * p1 / (apx + EPSV);
  float den = sqrtf(q0 * q0 + q1 * q1 + EPSV);

  out[gp] = w0 * q0 / den + w1 * q1 / den;
}

__global__ void init_red(float* red) {
  if (threadIdx.x < 3 * Bn) red[threadIdx.x] = 0.f;
}

// ---------------------------------------------------------------------------
extern "C" void kernel_launch(void* const* d_in, const int* in_sizes, int n_in,
                              void* d_out, int out_size, void* d_ws, size_t ws_size,
                              hipStream_t stream)
{
  const float* x      = (const float*)d_in[0];
  const float* W_in   = (const float*)d_in[1];
  const float* b_in   = (const float*)d_in[2];
  const float* W_gate = (const float*)d_in[3];
  const float* b_gate = (const float*)d_in[4];
  const float* W_t    = (const float*)d_in[5];
  const float* b_t    = (const float*)d_in[6];
  const float* conv_w = (const float*)d_in[7];
  const float* conv_b = (const float*)d_in[8];
  const float* wstat  = (const float*)d_in[9];
  const float* wmod   = (const float*)d_in[10];
  float* out = (float*)d_out;
  float* ws  = (float*)d_ws;

  // workspace layout (float units): ~38.3 MB total
  float*          xt     = ws + 0;        // [32, 8192] f32
  float*          xg     = ws + 262144;   // [32, 8192] f32
  unsigned short* ctxbf  = (unsigned short*)(ws + 524288); // [32, 8192] bf16
  unsigned short* xbf    = (unsigned short*)(ws + 655360); // [32, 1024] bf16
  float*          ctx2   = ws + 671744;   // [32, 8192] f32
  float*          modc   = ws + 933888;   // [32, 4, 4, 512] f32
  float*          ms     = ws + 1196032;  // [32, 256, 256, 2] f32
  float*          proj   = ws + 5390336;  // [32, 256, 256, 2] f32
  float*          red    = ws + 9584640;  // [3, 32] : max_ms, max_proj, sum_mag

  hipLaunchKernelGGL(init_red, dim3(1), dim3(128), 0, stream, red);
  hipLaunchKernelGGL(f32_to_bf16, dim3((Bn * CTXD + 255) / 256), dim3(256), 0, stream,
                     x, xbf, Bn * CTXD);

  // dual input GEMMs run concurrently via grid.y
  hipLaunchKernelGGL(gemm_wmma_bf16, dim3(FF / 64, 2), dim3(256), 0, stream,
                     W_in, b_in, xt, W_gate, b_gate, xg, xbf, FF, CTXD);

  hipLaunchKernelGGL(norm_gate, dim3(4, Bn), dim3(256), 0, stream, xt, xg, ctxbf);

  // big GEMM: ctx @ W_t.T  (streams 256 MB of W_t once)
  hipLaunchKernelGGL(gemm_wmma_bf16, dim3(FF / 64, 1), dim3(256), 0, stream,
                     W_t, b_t, ctx2, W_t, b_t, ctx2, ctxbf, FF, FF);

  hipLaunchKernelGGL(norm512, dim3(16, Bn), dim3(256), 0, stream, ctx2, modc);

  hipLaunchKernelGGL(pixel_pass1, dim3(Hd, Bn), dim3(256), 0, stream,
                     modc, conv_w, conv_b, wstat, wmod, ms, proj, red);
  hipLaunchKernelGGL(pixel_pass2, dim3(Hd, Bn), dim3(256), 0, stream, ms, red);
  hipLaunchKernelGGL(pixel_pass3, dim3(Hd, Bn), dim3(256), 0, stream,
                     ms, proj, wstat, red, out);
}
